// MultiHeadSelfAttention_13846974562946
// MI455X (gfx1250) — matrix operand order repaired and checked by reading, not compiled, not run on hardware
//
#include <hip/hip_runtime.h>
#include <hip/hip_bf16.h>

typedef _Float16 v16h __attribute__((ext_vector_type(16)));
typedef _Float16 v8h  __attribute__((ext_vector_type(8)));
typedef _Float16 v4h  __attribute__((ext_vector_type(4)));
typedef float    v8f  __attribute__((ext_vector_type(8)));
typedef float    v4f  __attribute__((ext_vector_type(4)));

#define D_EMB  1024
#define NHEADS 16
#define HDIM   64
#define BATCH  4
#define SEQ    2048
#define MROWS  (BATCH * SEQ)      // 8192
#define N_QKV  (3 * D_EMB)        // 3072

// ---------------------------------------------------------------------------
// WMMA fragment loaders (f16, 16x16x32). ISA layouts:
//  A 16x32: lane(idx=lane&15,half=lane>>4) holds row=idx,
//           elems 0..7 -> K = k0 + 8*half + e ; elems 8..15 -> K = k0+16+8*half+e
//  B 32x16: lane holds col=idx, same K order as A (elems 0..7 -> k0+8*half+e, 8..15 -> k0+16+8*half+e)
// ---------------------------------------------------------------------------
__device__ __forceinline__ v16h load_a_frag(const _Float16* base, int row, int ld, int k0) {
    const int lane = threadIdx.x & 31;
    const int half = lane >> 4;
    const _Float16* p = base + (size_t)row * ld + k0 + 8 * half;
    v8h lo = *(const v8h*)(p);
    v8h hi = *(const v8h*)(p + 16);
    return __builtin_shufflevector(lo, hi, 0,1,2,3,4,5,6,7,8,9,10,11,12,13,14,15);
}

__device__ __forceinline__ v16h load_b_frag(const _Float16* base, int col, int ld, int k0) {
    const int lane = threadIdx.x & 31;
    const int half = lane >> 4;
    const _Float16* p = base + (size_t)col * ld + k0 + 8 * half;
    v8h lo = *(const v8h*)(p);
    v8h hi = *(const v8h*)(p + 16);
    return __builtin_shufflevector(lo, hi, 0,1,2,3,4,5,6,7,8,9,10,11,12,13,14,15);
}

__device__ __forceinline__ v8f wmma_f16(v16h a, v16h b, v8f c) {
    return __builtin_amdgcn_wmma_f32_16x16x32_f16(false, a, false, b, (short)0, c, false, false);
}

// width-16 cross-lane reductions (rows of a C tile live within one 16-lane half)
__device__ __forceinline__ float red_max16(float x) {
#pragma unroll
    for (int off = 1; off < 16; off <<= 1) x = fmaxf(x, __shfl_xor(x, off, 16));
    return x;
}
__device__ __forceinline__ float red_sum16(float x) {
#pragma unroll
    for (int off = 1; off < 16; off <<= 1) x += __shfl_xor(x, off, 16);
    return x;
}

// ---------------------------------------------------------------------------
// fp32 -> f16 conversion, float4 vectorized (n4 = n/4)
// ---------------------------------------------------------------------------
__global__ void __launch_bounds__(256) cvt_f32_f16(const float* __restrict__ src,
                                                   _Float16* __restrict__ dst, int n4) {
    int i = blockIdx.x * blockDim.x + threadIdx.x;
    const int stride = gridDim.x * blockDim.x;
    for (; i < n4; i += stride) {
        v4f f = *(const v4f*)(src + 4 * (size_t)i);
        v4h h;
#pragma unroll
        for (int j = 0; j < 4; ++j) h[j] = (_Float16)f[j];
        *(v4h*)(dst + 4 * (size_t)i) = h;
    }
}

// ---------------------------------------------------------------------------
// GEMM building blocks: wave tile 32x64, ping-pong double buffering.
// ---------------------------------------------------------------------------
__device__ __forceinline__ void gemm_load(const _Float16* __restrict__ A,
                                          const _Float16* __restrict__ W,
                                          int m0, int n0, int idx, int k0,
                                          v16h (&a)[2], v16h (&b)[4]) {
#pragma unroll
    for (int r = 0; r < 2; ++r) a[r] = load_a_frag(A, m0 + 16 * r + idx, D_EMB, k0);
#pragma unroll
    for (int t = 0; t < 4; ++t) b[t] = load_b_frag(W, n0 + 16 * t + idx, D_EMB, k0);
}

__device__ __forceinline__ void gemm_step(v8f (&acc)[2][4], const v16h (&a)[2],
                                          const v16h (&b)[4]) {
#pragma unroll
    for (int r = 0; r < 2; ++r)
#pragma unroll
        for (int t = 0; t < 4; ++t)
            acc[r][t] = wmma_f16(a[r], b[t], acc[r][t]);
}

// K-loop: preload k=0 into A-buf; each unrolled-x2 iteration preloads the
// buffer not in use -> zero register copies, partial loadcnt waits.
__device__ __forceinline__ void gemm_mainloop(const _Float16* __restrict__ A,
                                              const _Float16* __restrict__ W,
                                              int m0, int n0, int idx,
                                              v8f (&acc)[2][4]) {
    v16h aA[2], bA[4], aB[2], bB[4];
    gemm_load(A, W, m0, n0, idx, 0, aA, bA);
    for (int k0 = 0; k0 < D_EMB - 64; k0 += 64) {
        gemm_load(A, W, m0, n0, idx, k0 + 32, aB, bB);
        gemm_step(acc, aA, bA);
        gemm_load(A, W, m0, n0, idx, k0 + 64, aA, bA);
        gemm_step(acc, aB, bB);
    }
    gemm_load(A, W, m0, n0, idx, D_EMB - 32, aB, bB);
    gemm_step(acc, aA, bA);
    gemm_step(acc, aB, bB);
}

// ---------------------------------------------------------------------------
// Kernel 1: QKV projection. C[m,n] = sum_k xh[m,k]*wqkv[n,k] + bias[n]
// block = 256 thr (8 waves). Wave tile 32x64, block tile 128x128.
// Scatters into Q[bh,s,64], K[bh,s,64], Vt[bh,64,s] (f16).
// ---------------------------------------------------------------------------
__global__ void __launch_bounds__(256) qkv_gemm_kernel(const _Float16* __restrict__ xh,
                                                       const _Float16* __restrict__ wh,
                                                       const float* __restrict__ bias,
                                                       _Float16* __restrict__ Qh,
                                                       _Float16* __restrict__ Kh,
                                                       _Float16* __restrict__ Vt) {
    const int wave = threadIdx.x >> 5;
    const int lane = threadIdx.x & 31;
    const int idx  = lane & 15;
    const int half = lane >> 4;
    const int m0 = blockIdx.x * 128 + (wave >> 1) * 32;
    const int n0 = blockIdx.y * 128 + (wave & 1) * 64;

    v8f acc[2][4];
#pragma unroll
    for (int r = 0; r < 2; ++r)
#pragma unroll
        for (int t = 0; t < 4; ++t)
#pragma unroll
            for (int i = 0; i < 8; ++i) acc[r][t][i] = 0.0f;

    gemm_mainloop(xh, wh, m0, n0, idx, acc);

#pragma unroll
    for (int t = 0; t < 4; ++t) {
        const int n = n0 + 16 * t + idx;
        const int h = n / (3 * HDIM);
        const int rr = n % (3 * HDIM);
        const float bv = bias[n];
#pragma unroll
        for (int r = 0; r < 2; ++r) {
#pragma unroll
            for (int v = 0; v < 8; ++v) {
                const int m  = m0 + 16 * r + v + 8 * half;
                const int bb = m >> 11;          // / SEQ
                const int s  = m & (SEQ - 1);
                const _Float16 val = (_Float16)(acc[r][t][v] + bv);
                const size_t bh = (size_t)(bb * NHEADS + h);
                if (rr < HDIM)           Qh[(bh * SEQ + s) * HDIM + rr] = val;
                else if (rr < 2 * HDIM)  Kh[(bh * SEQ + s) * HDIM + (rr - HDIM)] = val;
                else                     Vt[(bh * HDIM + (rr - 2 * HDIM)) * SEQ + s] = val;
            }
        }
    }
}

// ---------------------------------------------------------------------------
// Attention tile: one 32-key step. Computes scores with kb_use, prefetches
// next tile's K fragments into kb_load (ping-pong, no copies), online
// softmax, P.V accumulation. V loads issued before score WMMAs so their
// latency hides under softmax.
// ---------------------------------------------------------------------------
__device__ __forceinline__ void attn_tile(const _Float16* __restrict__ Kb,
                                          const _Float16* __restrict__ Vb,
                                          _Float16* pw,   // per-wave 32x32 LDS tile
                                          int idx, int half, int j0, int jn,
                                          const v16h (&qa)[2][2],
                                          v16h (&kb_use)[2][2], v16h (&kb_load)[2][2],
                                          v8f (&o)[2][4],
                                          float (&mrow)[2][8], float (&lrow)[2][8]) {
    const float scale = 0.125f;   // 1/sqrt(64)

    // V fragments for this tile (independent of scores -> issue first)
    v16h vb[4];
#pragma unroll
    for (int t = 0; t < 4; ++t) vb[t] = load_b_frag(Vb, 16 * t + idx, SEQ, j0);

    // scores: Q(32x64) . K^T(64x32) -> four 16x16 tiles
    v8f sc[2][2];
#pragma unroll
    for (int r = 0; r < 2; ++r)
#pragma unroll
        for (int t = 0; t < 2; ++t)
#pragma unroll
            for (int i = 0; i < 8; ++i) sc[r][t][i] = 0.0f;
#pragma unroll
    for (int r = 0; r < 2; ++r)
#pragma unroll
        for (int t = 0; t < 2; ++t)
#pragma unroll
            for (int kk = 0; kk < 2; ++kk)
                sc[r][t] = wmma_f16(qa[r][kk], kb_use[t][kk], sc[r][t]);

    // prefetch next tile's K fragments (latency hides under softmax)
#pragma unroll
    for (int t = 0; t < 2; ++t)
#pragma unroll
        for (int kk = 0; kk < 2; ++kk)
            kb_load[t][kk] = load_b_frag(Kb, jn + 16 * t + idx, HDIM, 32 * kk);

    // online softmax (per 16-row query group)
    float fac[2][8];
#pragma unroll
    for (int r = 0; r < 2; ++r) {
#pragma unroll
        for (int v = 0; v < 8; ++v) {
            const float s0 = sc[r][0][v] * scale;
            const float s1 = sc[r][1][v] * scale;
            const float mt = red_max16(fmaxf(s0, s1));
            const float mnew = fmaxf(mrow[r][v], mt);
            fac[r][v] = __expf(mrow[r][v] - mnew);
            mrow[r][v] = mnew;
            const float p0 = __expf(s0 - mnew);
            const float p1 = __expf(s1 - mnew);
            pw[(16 * r + v + 8 * half) * 32 + idx]      = (_Float16)p0;
            pw[(16 * r + v + 8 * half) * 32 + idx + 16] = (_Float16)p1;
            lrow[r][v] = lrow[r][v] * fac[r][v] + red_sum16(p0 + p1);
        }
    }

#pragma unroll
    for (int r = 0; r < 2; ++r)
#pragma unroll
        for (int t = 0; t < 4; ++t)
#pragma unroll
            for (int v = 0; v < 8; ++v) o[r][t][v] *= fac[r][v];

    // P (32x32) as A-fragments from LDS, P.V WMMAs
#pragma unroll
    for (int r = 0; r < 2; ++r) {
        const v16h pa = load_a_frag(pw, 16 * r + idx, 32, 0);
#pragma unroll
        for (int t = 0; t < 4; ++t)
            o[r][t] = wmma_f16(pa, vb[t], o[r][t]);
    }
}

// ---------------------------------------------------------------------------
// Kernel 2: flash attention per (b,h). block = 128 thr (4 waves),
// wave = 32 query rows, streaming 32-key tiles (j-loop unrolled x2 for
// ping-pong K-fragment prefetch).
// ---------------------------------------------------------------------------
__global__ void __launch_bounds__(128) attn_kernel(const _Float16* __restrict__ Qh,
                                                   const _Float16* __restrict__ Kh,
                                                   const _Float16* __restrict__ Vt,
                                                   _Float16* __restrict__ Oh) {
    __shared__ _Float16 plds[4][32][32];   // per-wave 32x32 P tile

    const int wave = threadIdx.x >> 5;
    const int lane = threadIdx.x & 31;
    const int idx  = lane & 15;
    const int half = lane >> 4;
    const int bh   = blockIdx.y;
    const int q0   = blockIdx.x * 128 + wave * 32;

    const _Float16* Qb = Qh + (size_t)bh * SEQ * HDIM;
    const _Float16* Kb = Kh + (size_t)bh * SEQ * HDIM;
    const _Float16* Vb = Vt + (size_t)bh * HDIM * SEQ;
    _Float16* pw = &plds[wave][0][0];

    v16h qa[2][2];
#pragma unroll
    for (int r = 0; r < 2; ++r)
#pragma unroll
        for (int kk = 0; kk < 2; ++kk)
            qa[r][kk] = load_a_frag(Qb, q0 + 16 * r + idx, HDIM, 32 * kk);

    v8f o[2][4];
    float mrow[2][8], lrow[2][8];
#pragma unroll
    for (int r = 0; r < 2; ++r) {
#pragma unroll
        for (int t = 0; t < 4; ++t)
#pragma unroll
            for (int i = 0; i < 8; ++i) o[r][t][i] = 0.0f;
#pragma unroll
        for (int v = 0; v < 8; ++v) { mrow[r][v] = -__builtin_inff(); lrow[r][v] = 0.0f; }
    }

    // K-fragment ping-pong: preload tile 0, each attn_tile call prefetches the
    // next tile into the other buffer. Final prefetch wraps (& SEQ-1), harmless.
    v16h kbA[2][2], kbB[2][2];
#pragma unroll
    for (int t = 0; t < 2; ++t)
#pragma unroll
        for (int kk = 0; kk < 2; ++kk)
            kbA[t][kk] = load_b_frag(Kb, 16 * t + idx, HDIM, 32 * kk);

    for (int jj = 0; jj < SEQ; jj += 64) {
        attn_tile(Kb, Vb, pw, idx, half, jj,      (jj + 32) & (SEQ - 1),
                  qa, kbA, kbB, o, mrow, lrow);
        attn_tile(Kb, Vb, pw, idx, half, jj + 32, (jj + 64) & (SEQ - 1),
                  qa, kbB, kbA, o, mrow, lrow);
    }

    const int bb = bh >> 4;
    const int h  = bh & 15;
#pragma unroll
    for (int r = 0; r < 2; ++r) {
#pragma unroll
        for (int t = 0; t < 4; ++t) {
#pragma unroll
            for (int v = 0; v < 8; ++v) {
                const int srow = q0 + 16 * r + v + 8 * half;
                const float val = o[r][t][v] / lrow[r][v];
                Oh[((size_t)(bb * SEQ + srow)) * D_EMB + h * HDIM + 16 * t + idx] = (_Float16)val;
            }
        }
    }
}

// ---------------------------------------------------------------------------
// Kernel 3: output projection. out[m,n] = sum_k Oh[m,k]*wout[n,k] + bias[n]
// ---------------------------------------------------------------------------
__global__ void __launch_bounds__(256) out_gemm_kernel(const _Float16* __restrict__ Oh,
                                                       const _Float16* __restrict__ wh,
                                                       const float* __restrict__ bias,
                                                       float* __restrict__ out) {
    const int wave = threadIdx.x >> 5;
    const int lane = threadIdx.x & 31;
    const int idx  = lane & 15;
    const int half = lane >> 4;
    const int m0 = blockIdx.x * 128 + (wave >> 1) * 32;
    const int n0 = blockIdx.y * 128 + (wave & 1) * 64;

    v8f acc[2][4];
#pragma unroll
    for (int r = 0; r < 2; ++r)
#pragma unroll
        for (int t = 0; t < 4; ++t)
#pragma unroll
            for (int i = 0; i < 8; ++i) acc[r][t][i] = 0.0f;

    gemm_mainloop(Oh, wh, m0, n0, idx, acc);

#pragma unroll
    for (int t = 0; t < 4; ++t) {
        const int n = n0 + 16 * t + idx;
        const float bv = bias[n];
#pragma unroll
        for (int r = 0; r < 2; ++r) {
#pragma unroll
            for (int v = 0; v < 8; ++v) {
                const int m = m0 + 16 * r + v + 8 * half;
                out[(size_t)m * D_EMB + n] = acc[r][t][v] + bv;
            }
        }
    }
}

// ---------------------------------------------------------------------------
extern "C" void kernel_launch(void* const* d_in, const int* in_sizes, int n_in,
                              void* d_out, int out_size, void* d_ws, size_t ws_size,
                              hipStream_t stream) {
    const float* x     = (const float*)d_in[0];
    const float* w_qkv = (const float*)d_in[1];
    const float* b_qkv = (const float*)d_in[2];
    const float* w_out = (const float*)d_in[3];
    const float* b_out = (const float*)d_in[4];
    float* out = (float*)d_out;

    // workspace carve (f16): xh | wqkvh | wouth | Q | K | Vt | Oh  (~88 MB)
    _Float16* xh    = (_Float16*)d_ws;
    _Float16* wqkvh = xh    + (size_t)MROWS * D_EMB;
    _Float16* wouth = wqkvh + (size_t)N_QKV * D_EMB;
    _Float16* Qh    = wouth + (size_t)D_EMB * D_EMB;
    _Float16* Kh    = Qh    + (size_t)MROWS * HDIM * NHEADS;
    _Float16* Vt    = Kh    + (size_t)MROWS * HDIM * NHEADS;
    _Float16* Oh    = Vt    + (size_t)MROWS * HDIM * NHEADS;

    cvt_f32_f16<<<2048, 256, 0, stream>>>(x,     xh,    (MROWS * D_EMB) / 4);
    cvt_f32_f16<<<2048, 256, 0, stream>>>(w_qkv, wqkvh, (N_QKV * D_EMB) / 4);
    cvt_f32_f16<<<1024, 256, 0, stream>>>(w_out, wouth, (D_EMB * D_EMB) / 4);

    qkv_gemm_kernel<<<dim3(MROWS / 128, N_QKV / 128), 256, 0, stream>>>(
        xh, wqkvh, b_qkv, Qh, Kh, Vt);

    attn_kernel<<<dim3(SEQ / 128, BATCH * NHEADS), 128, 0, stream>>>(Qh, Kh, Vt, Oh);

    out_gemm_kernel<<<dim3(MROWS / 128, D_EMB / 128), 256, 0, stream>>>(
        Oh, wouth, b_out, out);
}
